// RoPESelfAttention_17703855194211
// MI455X (gfx1250) — compile-verified, matrix operand order checked by reading, not run on hardware
//
#include <hip/hip_runtime.h>
#include <hip/hip_bf16.h>

typedef _Float16 half_t;
typedef __attribute__((ext_vector_type(16))) _Float16 v16h;
typedef __attribute__((ext_vector_type(8)))  float    v8f;

#define DIM        512
#define NUM_HEADS  8
#define HEAD_DIM   64
#define SEQ        4096
#define BATCH      2
#define MROWS      (BATCH * SEQ)   // 8192
#define QKV_N      (3 * DIM)       // 1536

__device__ __forceinline__ v8f wmma16(v16h a, v16h b, v8f c) {
    // D = A(16x32 f16) * B(32x16 f16) + C(16x16 f32)
    return __builtin_amdgcn_wmma_f32_16x16x32_f16(
        /*neg_a=*/false, a, /*neg_b=*/false, b,
        /*c_mod=*/(short)0, c, /*reuse_a=*/false, /*reuse_b=*/false);
}

// 16-bit operand K-mapping (CDNA5 ISA §7.12.2):
// lane group base = (lane>>4)*8 ; elem i<8 -> K=base+i ; elem i>=8 -> K=16+base+(i-8)
__device__ __forceinline__ int elemK(int lane, int i) {
    const int base = (lane >> 4) * 8;
    return (i < 8) ? (base + i) : (16 + base + (i - 8));
}

// ---------------------------------------------------------------------------
// Kernel 0: pack a f32 row-major weight W[K][N] into f16 "fragment-linear"
// layout: Wh[((ntile*(K/32)+kc)*32 + lane)*16 + i] = W[kc*32+elemK(lane,i)][ntile*16+lane%16]
// so a WMMA B-fragment is one contiguous, 32-byte aligned v16h per lane.
// ---------------------------------------------------------------------------
__global__ void __launch_bounds__(256)
pack_w_kernel(const float* __restrict__ W, half_t* __restrict__ Wh, int K, int N) {
    const int idx = blockIdx.x * 256 + threadIdx.x;
    if (idx >= K * N) return;
    const int i       = idx & 15;
    const int lane    = (idx >> 4) & 31;
    const int chunk   = idx >> 9;
    const int nchunks = K >> 5;
    const int ntile   = chunk / nchunks;
    const int kc      = chunk - ntile * nchunks;
    const int k = kc * 32 + elemK(lane, i);
    const int n = ntile * 16 + (lane & 15);
    Wh[idx] = (half_t)W[(size_t)k * N + n];
}

// ---------------------------------------------------------------------------
// Kernel 1: qkv = x @ Wqkv + bqkv, RoPE on q,k.
// Q,K stored f16 [B,H,L,64]; V stored TRANSPOSED f16 [B,H,64,L].
// grid (512, 6), block 128 (4 waves). Wave tile: 16 rows x 64 cols (1 head).
// ---------------------------------------------------------------------------
__global__ void __launch_bounds__(128)
qkv_rope_kernel(const float* __restrict__ x, const half_t* __restrict__ WqkvH,
                const float* __restrict__ bqkv,
                const float* __restrict__ cosT, const float* __restrict__ sinT,
                half_t* __restrict__ Qh, half_t* __restrict__ Kh,
                half_t* __restrict__ Vt) {
    const int lane   = threadIdx.x & 31;
    const int wave   = threadIdx.x >> 5;
    const int row0   = blockIdx.x * 16;                 // M tile
    const int nstrip = blockIdx.y * 4 + wave;           // 64-col head-aligned strip
    const int ncol   = nstrip * 64;
    const int base   = (lane >> 4) * 8;
    const int nloc   = lane & 15;
    const int rhalf  = (lane >> 4) << 3;
    const int NCH    = DIM / 32;                        // k-chunks

    v8f acc[4] = {};

    for (int kc = 0; kc < NCH; ++kc) {
        // A fragment: 16 rows of x (f32 -> f16)
        const float* ap = x + (size_t)(row0 + nloc) * DIM + kc * 32;
        v16h afrag;
#pragma unroll
        for (int i = 0; i < 8; ++i) {
            afrag[i]     = (half_t)ap[base + i];
            afrag[i + 8] = (half_t)ap[16 + base + i];
        }
#pragma unroll
        for (int t = 0; t < 4; ++t) {
            const int ntile = nstrip * 4 + t;
            const v16h bfrag = *(const v16h*)(WqkvH +
                ((size_t)(ntile * NCH + kc) * 32 + lane) * 16);
            acc[t] = wmma16(afrag, bfrag, acc[t]);
        }
    }

    // bias
#pragma unroll
    for (int t = 0; t < 4; ++t) {
        const float bv = bqkv[ncol + t * 16 + nloc];
#pragma unroll
        for (int r = 0; r < 8; ++r) acc[t][r] += bv;
    }

    // RoPE for q (ncol<512) and k (512<=ncol<1024): pair (acc[t], acc[t+2])
    if (ncol < 2 * DIM) {
#pragma unroll
        for (int pr = 0; pr < 2; ++pr) {
            const int d0 = pr * 16 + nloc;   // in-head dim 0..31
            const int d1 = d0 + 32;
#pragma unroll
            for (int r = 0; r < 8; ++r) {
                const int m   = row0 + r + rhalf;
                const int pos = m & (SEQ - 1);
                const float c0 = cosT[pos * HEAD_DIM + d0];
                const float s0 = sinT[pos * HEAD_DIM + d0];
                const float c1 = cosT[pos * HEAD_DIM + d1];
                const float s1 = sinT[pos * HEAD_DIM + d1];
                const float lo = acc[pr][r], hi = acc[pr + 2][r];
                acc[pr][r]     = lo * c0 - hi * s0;  // d<32: rot = -q[d+32]
                acc[pr + 2][r] = hi * c1 + lo * s1;  // d>=32: rot = +q[d-32]
            }
        }
    }

    // store; Q pre-scaled by 1/sqrt(HEAD_DIM); V goes out transposed
    const float oscale = (ncol < DIM) ? 0.125f : 1.0f;
#pragma unroll
    for (int t = 0; t < 4; ++t) {
        const int n     = ncol + t * 16 + nloc;
        const int which = n >> 9;              // 0=q 1=k 2=v
        const int h     = (n & 511) >> 6;
        const int d     = n & 63;
#pragma unroll
        for (int r = 0; r < 8; ++r) {
            const int m   = row0 + r + rhalf;
            const int b   = m >> 12;
            const int pos = m & (SEQ - 1);
            const int bh  = b * NUM_HEADS + h;
            const half_t v = (half_t)(acc[t][r] * oscale);
            if (which == 0)
                Qh[((size_t)bh * SEQ + pos) * HEAD_DIM + d] = v;
            else if (which == 1)
                Kh[((size_t)bh * SEQ + pos) * HEAD_DIM + d] = v;
            else
                Vt[((size_t)bh * HEAD_DIM + d) * SEQ + pos] = v;   // [B,H,d,L]
        }
    }
}

// ---------------------------------------------------------------------------
// Kernel 2: flash attention. grid (32, 16=B*H), block 256 (8 waves).
// Wave: 16 queries, online softmax over 32 keys/iter. O stored f16 [B,L,512].
// ---------------------------------------------------------------------------
__global__ void __launch_bounds__(256)
attention_kernel(const half_t* __restrict__ Qh, const half_t* __restrict__ Kh,
                 const half_t* __restrict__ Vt, const int* __restrict__ pad,
                 half_t* __restrict__ Oh) {
    __shared__ half_t plds[8][16 * 32];   // per-wave private P-transpose patch

    const int lane  = threadIdx.x & 31;
    const int wave  = threadIdx.x >> 5;
    const int bh    = blockIdx.y;                 // b*8+h
    const int b     = bh >> 3;
    const int h     = bh & 7;
    const int qbase = blockIdx.x * 128 + wave * 16;
    const int base  = (lane >> 4) * 8;
    const int nloc  = lane & 15;
    const int rhalf = (lane >> 4) << 3;

    const half_t* Qp = Qh + (size_t)bh * SEQ * HEAD_DIM;
    const half_t* Kp = Kh + (size_t)bh * SEQ * HEAD_DIM;
    const half_t* Vp = Vt + (size_t)bh * HEAD_DIM * SEQ;
    const int*    pm = pad + b * SEQ;

    // Q A-fragments for d=0..31 and d=32..63 (held for whole key loop)
    v16h qa0, qa1;
    {
        const half_t* qp = Qp + (size_t)(qbase + nloc) * HEAD_DIM;
#pragma unroll
        for (int i = 0; i < 8; ++i) {
            qa0[i]     = qp[base + i];
            qa0[i + 8] = qp[16 + base + i];
            qa1[i]     = qp[32 + base + i];
            qa1[i + 8] = qp[48 + base + i];
        }
    }

    v8f oac[4] = {};
    float mrow[8], lsum[8];
#pragma unroll
    for (int r = 0; r < 8; ++r) { mrow[r] = -INFINITY; lsum[r] = 0.f; }

    half_t* pl = plds[wave];

    for (int kt = 0; kt < SEQ; kt += 32) {
        // ---- scores for two 16-key tiles: S = (Q/8) @ K^T ----
        v16h kb00, kb01, kb10, kb11;
        {
            const half_t* kp0 = Kp + (size_t)(kt + nloc) * HEAD_DIM;
            const half_t* kp1 = Kp + (size_t)(kt + 16 + nloc) * HEAD_DIM;
#pragma unroll
            for (int i = 0; i < 8; ++i) {
                kb00[i]     = kp0[base + i];      kb00[i + 8] = kp0[16 + base + i];
                kb01[i]     = kp0[32 + base + i]; kb01[i + 8] = kp0[48 + base + i];
                kb10[i]     = kp1[base + i];      kb10[i + 8] = kp1[16 + base + i];
                kb11[i]     = kp1[32 + base + i]; kb11[i + 8] = kp1[48 + base + i];
            }
        }
        v8f s0 = {}, s1 = {};
        s0 = wmma16(qa0, kb00, s0);
        s0 = wmma16(qa1, kb01, s0);
        s1 = wmma16(qa0, kb10, s1);
        s1 = wmma16(qa1, kb11, s1);

        // ---- key padding mask (key index = lane column) ----
        if (pm[kt + nloc]) {
#pragma unroll
            for (int r = 0; r < 8; ++r) s0[r] = -INFINITY;
        }
        if (pm[kt + 16 + nloc]) {
#pragma unroll
            for (int r = 0; r < 8; ++r) s1[r] = -INFINITY;
        }

        // ---- online softmax: row max across the 16 lanes of each half ----
        float mx[8];
#pragma unroll
        for (int r = 0; r < 8; ++r) mx[r] = fmaxf(s0[r], s1[r]);
#pragma unroll
        for (int off = 8; off > 0; off >>= 1)
#pragma unroll
            for (int r = 0; r < 8; ++r) mx[r] = fmaxf(mx[r], __shfl_xor(mx[r], off, 32));

        float p0[8], p1[8], alpha[8];
#pragma unroll
        for (int r = 0; r < 8; ++r) {
            const float nm = fmaxf(mrow[r], mx[r]);
            if (nm == -INFINITY) { alpha[r] = 1.f; p0[r] = 0.f; p1[r] = 0.f; }
            else {
                alpha[r] = __expf(mrow[r] - nm);
                p0[r] = (s0[r] == -INFINITY) ? 0.f : __expf(s0[r] - nm);
                p1[r] = (s1[r] == -INFINITY) ? 0.f : __expf(s1[r] - nm);
            }
            mrow[r] = nm;
        }

        float rs[8];
#pragma unroll
        for (int r = 0; r < 8; ++r) rs[r] = p0[r] + p1[r];
#pragma unroll
        for (int off = 8; off > 0; off >>= 1)
#pragma unroll
            for (int r = 0; r < 8; ++r) rs[r] += __shfl_xor(rs[r], off, 32);
#pragma unroll
        for (int r = 0; r < 8; ++r) lsum[r] = lsum[r] * alpha[r] + rs[r];
#pragma unroll
        for (int t = 0; t < 4; ++t)
#pragma unroll
            for (int r = 0; r < 8; ++r) oac[t][r] *= alpha[r];

        // ---- transpose P (C-layout -> A-layout) through private LDS ----
#pragma unroll
        for (int r = 0; r < 8; ++r) {
            const int row = r + rhalf;
            pl[row * 32 + nloc]      = (half_t)p0[r];
            pl[row * 32 + 16 + nloc] = (half_t)p1[r];
        }
        asm volatile("s_wait_dscnt 0x0" ::: "memory");   // same-wave DS RAW

        v16h pa;
        {
            const half_t* pr_ = pl + nloc * 32;
#pragma unroll
            for (int i = 0; i < 8; ++i) {
                pa[i]     = pr_[base + i];
                pa[i + 8] = pr_[16 + base + i];
            }
        }

        // ---- O += P @ V : 4 d-tiles, V pre-transposed -> contiguous runs ----
#pragma unroll
        for (int t = 0; t < 4; ++t) {
            const int d = t * 16 + nloc;
            const half_t* vp = Vp + (size_t)d * SEQ + kt;
            v16h vb;
#pragma unroll
            for (int i = 0; i < 8; ++i) {
                vb[i]     = vp[base + i];
                vb[i + 8] = vp[16 + base + i];
            }
            oac[t] = wmma16(pa, vb, oac[t]);
        }
    }

    // ---- normalize and write O as [B,L,512] f16 (row-major for proj A-frags) ----
#pragma unroll
    for (int t = 0; t < 4; ++t) {
        const int d = t * 16 + nloc;
#pragma unroll
        for (int r = 0; r < 8; ++r) {
            const int q   = qbase + r + rhalf;
            const float inv = (lsum[r] > 0.f) ? (1.f / lsum[r]) : 0.f;
            Oh[(size_t)(b * SEQ + q) * DIM + h * HEAD_DIM + d] =
                (half_t)(oac[t][r] * inv);
        }
    }
}

// ---------------------------------------------------------------------------
// Kernel 3: y = O @ Wproj + bproj + x ; LayerNorm(y). grid 512, block 256.
// Block: 16 rows x 512 cols, wave w owns cols [64w, 64w+64).
// ---------------------------------------------------------------------------
__global__ void __launch_bounds__(256)
proj_ln_kernel(const half_t* __restrict__ Oh, const half_t* __restrict__ WprojH,
               const float* __restrict__ bproj, const float* __restrict__ x,
               const float* __restrict__ gamma, const float* __restrict__ beta,
               float* __restrict__ out) {
    __shared__ float yt[16][DIM];

    const int lane  = threadIdx.x & 31;
    const int wave  = threadIdx.x >> 5;
    const int row0  = blockIdx.x * 16;
    const int ncol  = wave * 64;
    const int base  = (lane >> 4) * 8;
    const int nloc  = lane & 15;
    const int rhalf = (lane >> 4) << 3;
    const int NCH   = DIM / 32;

    v8f acc[4] = {};
    for (int kc = 0; kc < NCH; ++kc) {
        const half_t* ap = Oh + (size_t)(row0 + nloc) * DIM + kc * 32;
        v16h afrag;
#pragma unroll
        for (int i = 0; i < 8; ++i) {
            afrag[i]     = ap[base + i];
            afrag[i + 8] = ap[16 + base + i];
        }
#pragma unroll
        for (int t = 0; t < 4; ++t) {
            const int ntile = wave * 4 + t;
            const v16h bfrag = *(const v16h*)(WprojH +
                ((size_t)(ntile * NCH + kc) * 32 + lane) * 16);
            acc[t] = wmma16(afrag, bfrag, acc[t]);
        }
    }

    // bias + residual -> LDS tile
#pragma unroll
    for (int t = 0; t < 4; ++t) {
        const int n = ncol + t * 16 + nloc;
#pragma unroll
        for (int r = 0; r < 8; ++r) {
            const int rl = r + rhalf;
            yt[rl][n] = acc[t][r] + bproj[n] + x[(size_t)(row0 + rl) * DIM + n];
        }
    }
    __syncthreads();

    // LayerNorm: wave handles rows 2w and 2w+1
#pragma unroll
    for (int rr = 0; rr < 2; ++rr) {
        const int rl = wave * 2 + rr;
        float s = 0.f, sq = 0.f;
        for (int c = lane; c < DIM; c += 32) {
            const float v = yt[rl][c];
            s += v; sq += v * v;
        }
#pragma unroll
        for (int off = 16; off > 0; off >>= 1) {
            s  += __shfl_xor(s,  off, 32);
            sq += __shfl_xor(sq, off, 32);
        }
        const float mu  = s * (1.0f / DIM);
        const float var = sq * (1.0f / DIM) - mu * mu;
        const float rsd = rsqrtf(var + 1e-5f);
        const size_t rowoff = (size_t)(row0 + rl) * DIM;
        for (int c = lane; c < DIM; c += 32)
            out[rowoff + c] = (yt[rl][c] - mu) * rsd * gamma[c] + beta[c];
    }
}

// ---------------------------------------------------------------------------
extern "C" void kernel_launch(void* const* d_in, const int* in_sizes, int n_in,
                              void* d_out, int out_size, void* d_ws, size_t ws_size,
                              hipStream_t stream) {
    const float* x     = (const float*)d_in[0];
    const int*   pad   = (const int*)d_in[1];
    const float* Wqkv  = (const float*)d_in[2];
    const float* bqkv  = (const float*)d_in[3];
    const float* Wproj = (const float*)d_in[4];
    const float* bproj = (const float*)d_in[5];
    const float* gamma = (const float*)d_in[6];
    const float* beta  = (const float*)d_in[7];
    const float* cosT  = (const float*)d_in[8];
    const float* sinT  = (const float*)d_in[9];

    char* ws = (char*)d_ws;
    half_t* Qh     = (half_t*)(ws);                        // 8 MB
    half_t* Kh     = (half_t*)(ws + ((size_t)8  << 20));   // 8 MB
    half_t* Vt     = (half_t*)(ws + ((size_t)16 << 20));   // 8 MB (transposed)
    half_t* Ohalf  = (half_t*)(ws + ((size_t)24 << 20));   // 8 MB
    half_t* WqkvH  = (half_t*)(ws + ((size_t)32 << 20));   // 1.5 MB packed
    half_t* WprojH = (half_t*)(ws + ((size_t)34 << 20));   // 0.5 MB packed

    // pack weights to WMMA-fragment-linear f16
    pack_w_kernel<<<(DIM * QKV_N + 255) / 256, 256, 0, stream>>>(Wqkv, WqkvH, DIM, QKV_N);
    pack_w_kernel<<<(DIM * DIM + 255) / 256, 256, 0, stream>>>(Wproj, WprojH, DIM, DIM);

    dim3 g1(MROWS / 16, QKV_N / 256);   // 512 x 6
    qkv_rope_kernel<<<g1, 128, 0, stream>>>(x, WqkvH, bqkv, cosT, sinT, Qh, Kh, Vt);

    dim3 g2(SEQ / 128, BATCH * NUM_HEADS);  // 32 x 16
    attention_kernel<<<g2, 256, 0, stream>>>(Qh, Kh, Vt, pad, Ohalf);

    proj_ln_kernel<<<MROWS / 16, 256, 0, stream>>>(Ohalf, WprojH, bproj, x, gamma, beta,
                                                   (float*)d_out);
}
